// graphNet_4449586119110
// MI455X (gfx1250) — compile-verified, matrix operand order checked by reading, not run on hardware
//
#include <hip/hip_runtime.h>
#include <hip/hip_bf16.h>
#include <math.h>

#define HID 128
#define NWAVES 4
#define EPS_LN 1e-5f

typedef __attribute__((ext_vector_type(16))) _Float16 v16h;
typedef __attribute__((ext_vector_type(8)))  _Float16 v8h;
typedef __attribute__((ext_vector_type(4)))  _Float16 v4h;
typedef __attribute__((ext_vector_type(8)))  float    v8f;
typedef __attribute__((ext_vector_type(4)))  float    v4f;

enum { M_ENC = 0, M_EBLK = 1, M_NBLK = 2, M_DEC = 3 };

// ---------------------------------------------------------------------------
// Fragment loader: halves p[0..7] and p[16..23] as two b128 loads, matching the
// CDNA5 16-bit A-matrix 16x32 layout (row per lane, two 8-half K groups).
// ---------------------------------------------------------------------------
__device__ __forceinline__ v16h frag16(const _Float16* p) {
  v8h lo = *(const v8h*)(p);
  v8h hi = *(const v8h*)(p + 16);
  return __builtin_shufflevector(lo, hi, 0,1,2,3,4,5,6,7,8,9,10,11,12,13,14,15);
}

// Zero accumulators: compiler materializes the first WMMA's C operand as the
// inline constant 0, keeping acc VGPRs dead until their first WMMA -> deeper
// B-fragment load prefetch (s_clause) by the scheduler.
__device__ __forceinline__ void zero8(v8f* a) {
  #pragma unroll
  for (int i = 0; i < 8; ++i)
    #pragma unroll
    for (int j = 0; j < 8; ++j) a[i][j] = 0.f;
}

// Single-instruction ReLU: v_med3_f32(x, 0, +inf) == clamp to [0, inf).
__device__ __forceinline__ float relu1(float x) {
  return __builtin_amdgcn_fmed3f(x, 0.f, __builtin_inff());
}

// One dense layer: X[16][KPAD] (LDS, f16) x Wt[NT*16][KPAD] (global, f16) ->
// acc[NT] f32 fragments via v_wmma_f32_16x16x32_f16.
template<int KPAD, int NT>
__device__ __forceinline__ void wmma_layer(const _Float16* act, const _Float16* __restrict__ Wt,
                                           v8f* acc, int lane) {
  const int mr = lane & 15;          // row (A) / col (B) within tile
  const int kh = (lane >> 4) << 3;   // 8-half K offset for upper half-wave
  #pragma unroll
  for (int kt = 0; kt < KPAD / 32; ++kt) {
    v16h a = frag16(act + mr * KPAD + kt * 32 + kh);
    #pragma unroll
    for (int nt = 0; nt < NT; ++nt) {
      v16h b = frag16(Wt + (size_t)(nt * 16 + mr) * KPAD + kt * 32 + kh);
      acc[nt] = __builtin_amdgcn_wmma_f32_16x16x32_f16(
          /*neg_a=*/false, a, /*neg_b=*/false, b,
          /*c_mod=*/(short)0, acc[nt], /*reuse_a=*/false, /*reuse_b=*/false);
    }
  }
}

// Write 16x128 f32 fragments back to LDS as f16 with bias + ReLU (stride HID).
__device__ __forceinline__ void store_hidden_relu(_Float16* act, const v8f* acc,
                                                  const float* __restrict__ bias, int lane) {
  const int n  = lane & 15;
  const int mb = (lane >> 4) << 3;
  #pragma unroll
  for (int nt = 0; nt < 8; ++nt) {
    float bz = bias[nt * 16 + n];
    #pragma unroll
    for (int r = 0; r < 8; ++r)
      act[(mb + r) * HID + nt * 16 + n] = (_Float16)relu1(acc[nt][r] + bz);
  }
}

// Stage one 128-wide f32 row into LDS as f16: one b128 load + one b64 store
// per lane (lane owns 4 consecutive columns).
__device__ __forceinline__ void stage_row128(_Float16* dst, const float* __restrict__ src,
                                             int lane) {
  v4f v = *(const v4f*)(src + lane * 4);
  v4h h;
  #pragma unroll
  for (int j = 0; j < 4; ++j) h[j] = (_Float16)v[j];
  *(v4h*)(dst + lane * 4) = h;
}

// ---------------------------------------------------------------------------
// Fused row-MLP kernel. One wave = one 16-row tile.
//   M_ENC : X = f32 in0[rows][16] (zero-padded to K=32) -> MLP -> LN -> out
//   M_EBLK: X = [h_n[send], h_n[recv], h_e] (K=384) -> MLP -> LN -> +h_e
//   M_NBLK: X = [h_n, agg] (K=256) -> MLP -> LN -> +h_n
//   M_DEC : X = h_n (K=128) -> MLP -> out[rows][3]
// ---------------------------------------------------------------------------
template<int MODE>
__global__ void __launch_bounds__(NWAVES * 32)
k_mlp(const float* __restrict__ in0, const float* __restrict__ in1,
      const int* __restrict__ idxA, const int* __restrict__ idxB,
      const _Float16* __restrict__ W0t, const _Float16* __restrict__ W1t,
      const _Float16* __restrict__ W2t,
      const float* __restrict__ b0, const float* __restrict__ b1,
      const float* __restrict__ b2,
      const float* __restrict__ g, const float* __restrict__ beta,
      float* __restrict__ out, int rows) {
  constexpr int  KPAD  = (MODE == M_ENC) ? 32 : (MODE == M_EBLK) ? 384
                        : (MODE == M_NBLK) ? 256 : 128;
  constexpr bool LN    = (MODE != M_DEC);
  constexpr bool RESID = (MODE == M_EBLK || MODE == M_NBLK);
  // act tile reused as f32 [16][128] staging for LN -> needs 4096 halves.
  constexpr int  ACT_H = (LN && 16 * HID * 2 > 16 * KPAD) ? 16 * HID * 2 : 16 * KPAD;

  __shared__ alignas(16) _Float16 smem[NWAVES * ACT_H];

  const int lane = threadIdx.x & 31;
  const int wv   = threadIdx.x >> 5;
  const int tile = blockIdx.x * NWAVES + wv;
  if (tile * 16 >= rows) return;
  _Float16* act = smem + wv * ACT_H;
  const int r0 = tile * 16;

  // ---- input assembly: gather + concat + f32->f16 into LDS ----------------
  if constexpr (MODE == M_ENC) {
    #pragma unroll
    for (int r = 0; r < 16; ++r) {
      float v = (lane < 16) ? in0[(size_t)(r0 + r) * 16 + lane] : 0.f;
      act[r * 32 + lane] = (_Float16)v;   // lanes 16..31 write K-pad zeros
    }
  } else if constexpr (MODE == M_EBLK) {
    for (int r = 0; r < 16; ++r) {
      const int e = r0 + r;
      stage_row128(act + r * KPAD,           in0 + (size_t)idxA[e] * HID, lane);  // sender
      stage_row128(act + r * KPAD + HID,     in0 + (size_t)idxB[e] * HID, lane);  // receiver
      stage_row128(act + r * KPAD + 2 * HID, in1 + (size_t)e * HID,       lane);  // edge
    }
  } else if constexpr (MODE == M_NBLK) {
    for (int r = 0; r < 16; ++r) {
      const int n = r0 + r;
      stage_row128(act + r * KPAD,       in0 + (size_t)n * HID, lane);  // h_n
      stage_row128(act + r * KPAD + HID, in1 + (size_t)n * HID, lane);  // agg
    }
  } else {  // M_DEC
    for (int r = 0; r < 16; ++r)
      stage_row128(act + r * KPAD, in0 + (size_t)(r0 + r) * HID, lane);
  }
  asm volatile("s_wait_dscnt 0" ::: "memory");

  v8f acc[8];

  // layer 1: K=KPAD -> 128, ReLU
  zero8(acc);
  wmma_layer<KPAD, 8>(act, W0t, acc, lane);
  store_hidden_relu(act, acc, b0, lane);
  asm volatile("s_wait_dscnt 0" ::: "memory");

  // layer 2: 128 -> 128, ReLU
  zero8(acc);
  wmma_layer<HID, 8>(act, W1t, acc, lane);
  store_hidden_relu(act, acc, b1, lane);
  asm volatile("s_wait_dscnt 0" ::: "memory");

  if constexpr (MODE == M_DEC) {
    // layer 3: 128 -> 3 (one 16-col tile, cols 3..15 are zero weights)
    const int n  = lane & 15;
    const int mb = (lane >> 4) << 3;
    v8f a3;
    #pragma unroll
    for (int j = 0; j < 8; ++j) a3[j] = 0.f;
    wmma_layer<HID, 1>(act, W2t, &a3, lane);
    if (n < 3) {
      float bz = b2[n];
      #pragma unroll
      for (int r = 0; r < 8; ++r)
        out[(size_t)(r0 + mb + r) * 3 + n] = a3[r] + bz;
    }
  } else {
    // layer 3: 128 -> 128, then LayerNorm (+residual) in f32
    zero8(acc);
    wmma_layer<HID, 8>(act, W2t, acc, lane);
    float* fb = (float*)act;             // reuse act tile as f32 [16][128]
    const int n  = lane & 15;
    const int mb = (lane >> 4) << 3;
    #pragma unroll
    for (int nt = 0; nt < 8; ++nt) {
      float bz = b2[nt * 16 + n];
      #pragma unroll
      for (int r = 0; r < 8; ++r)
        fb[(mb + r) * HID + nt * 16 + n] = acc[nt][r] + bz;
    }
    asm volatile("s_wait_dscnt 0" ::: "memory");

    // lane handles (row = lane&15, half = lane>>4); pairwise combine via xor-16
    const int row = lane & 15;
    const int hh  = lane >> 4;
    const float* fr = fb + row * HID + hh * 64;
    float s = 0.f, s2 = 0.f;
    #pragma unroll
    for (int j = 0; j < 64; ++j) { float v = fr[j]; s += v; s2 += v * v; }
    s  += __shfl_xor(s, 16);
    s2 += __shfl_xor(s2, 16);
    const float mu   = s * (1.f / 128.f);
    const float rstd = rsqrtf(s2 * (1.f / 128.f) - mu * mu + EPS_LN);
    const size_t orow = (size_t)(r0 + row) * HID;
    #pragma unroll
    for (int j = 0; j < 64; ++j) {
      const int c = hh * 64 + j;
      float y = (fr[j] - mu) * rstd * g[c] + beta[c];
      if constexpr (RESID) y += out[orow + c];  // out still holds pre-update h
      out[orow + c] = y;
    }
  }
}

// ---------------------------------------------------------------------------
// Helpers: weight transpose f32[K][N] -> f16 Wt[NP][KP] (zero-padded),
// zero-fill, and atomic scatter-sum of edge features to receiver nodes.
// ---------------------------------------------------------------------------
__global__ void k_cvt_wt(const float* __restrict__ src, _Float16* __restrict__ dst,
                         int K, int N, int KP, int NP) {
  int i = blockIdx.x * blockDim.x + threadIdx.x;
  if (i >= KP * NP) return;
  int k = i % KP, n = i / KP;
  dst[i] = (k < K && n < N) ? (_Float16)src[(size_t)k * N + n] : (_Float16)0.f;
}

__global__ void k_zero(float* __restrict__ p, int n) {
  int i = blockIdx.x * blockDim.x + threadIdx.x;
  if (i < n) p[i] = 0.f;
}

__global__ void k_scatter(const float* __restrict__ he, const int* __restrict__ recv,
                          float* __restrict__ agg, int n_edges) {
  int i = blockIdx.x * blockDim.x + threadIdx.x;
  if (i >= n_edges * HID) return;
  int e = i >> 7, c = i & 127;
  atomicAdd(agg + (size_t)recv[e] * HID + c, he[i]);
}

// ---------------------------------------------------------------------------
// Host launcher.
// Input order (JAX tree_leaves: dict keys sorted, lists in order, {"W","b"},
// {"beta","g","mlp"} sorted):
//   0 nodes, 1 edges, 2 senders, 3 receivers,
//   4..9   decoder: W0,b0,W1,b1,W2,b2
//   10+8i  edge_blks[i]: beta,g,W0,b0,W1,b1,W2,b2   (i=0..3)
//   42..49 edge_enc:     beta,g,W0,b0,W1,b1,W2,b2
//   50+8i  node_blks[i]: beta,g,W0,b0,W1,b1,W2,b2   (i=0..3)
//   82..89 node_enc:     beta,g,W0,b0,W1,b1,W2,b2
// ---------------------------------------------------------------------------
extern "C" void kernel_launch(void* const* d_in, const int* in_sizes, int n_in,
                              void* d_out, int out_size, void* d_ws, size_t ws_size,
                              hipStream_t stream) {
  (void)n_in; (void)out_size; (void)ws_size;
  auto F = [&](int i) { return (const float*)d_in[i]; };

  const float* nodes     = F(0);
  const float* edges     = F(1);
  const int*   senders   = (const int*)d_in[2];
  const int*   receivers = (const int*)d_in[3];
  const int n_nodes = in_sizes[0] / 16;
  const int n_edges = in_sizes[1] / 16;

  // decoder
  const float *dW[3] = {F(4), F(6), F(8)};
  const float *dB[3] = {F(5), F(7), F(9)};
  // edge blocks
  const float *ebBeta[4], *ebG[4], *ebW[4][3], *ebB[4][3];
  for (int i = 0; i < 4; ++i) {
    int b = 10 + 8 * i;
    ebBeta[i] = F(b); ebG[i] = F(b + 1);
    ebW[i][0] = F(b + 2); ebB[i][0] = F(b + 3);
    ebW[i][1] = F(b + 4); ebB[i][1] = F(b + 5);
    ebW[i][2] = F(b + 6); ebB[i][2] = F(b + 7);
  }
  // edge encoder
  const float *eeBeta = F(42), *eeG = F(43);
  const float *eeW[3] = {F(44), F(46), F(48)};
  const float *eeB[3] = {F(45), F(47), F(49)};
  // node blocks
  const float *nbBeta[4], *nbG[4], *nbW[4][3], *nbB[4][3];
  for (int i = 0; i < 4; ++i) {
    int b = 50 + 8 * i;
    nbBeta[i] = F(b); nbG[i] = F(b + 1);
    nbW[i][0] = F(b + 2); nbB[i][0] = F(b + 3);
    nbW[i][1] = F(b + 4); nbB[i][1] = F(b + 5);
    nbW[i][2] = F(b + 6); nbB[i][2] = F(b + 7);
  }
  // node encoder
  const float *neBeta = F(82), *neG = F(83);
  const float *neW[3] = {F(84), F(86), F(88)};
  const float *neB[3] = {F(85), F(87), F(89)};

  // workspace carve: h_n, h_e, agg (f32) + f16 transposed weights
  char* base = (char*)d_ws;
  size_t off = 0;
  auto carve = [&](size_t bytes) -> void* {
    void* p = base + off;
    off += (bytes + 255) & ~(size_t)255;
    return p;
  };
  float* hn  = (float*)carve((size_t)n_nodes * HID * 4);
  float* he  = (float*)carve((size_t)n_edges * HID * 4);
  float* agg = (float*)carve((size_t)n_nodes * HID * 4);
  _Float16* wcur = (_Float16*)carve(2u * 1024u * 1024u);  // ~1.4MB of f16 weights

  auto cvt = [&](const float* src, int K, int N, int KP, int NP) -> const _Float16* {
    _Float16* dst = wcur;
    wcur += (size_t)KP * NP;
    int tot = KP * NP;
    k_cvt_wt<<<(tot + 255) / 256, 256, 0, stream>>>(src, dst, K, N, KP, NP);
    return dst;
  };

  const _Float16 *neWt[3] = {cvt(neW[0], 16, 128, 32, 128),
                             cvt(neW[1], 128, 128, 128, 128),
                             cvt(neW[2], 128, 128, 128, 128)};
  const _Float16 *eeWt[3] = {cvt(eeW[0], 16, 128, 32, 128),
                             cvt(eeW[1], 128, 128, 128, 128),
                             cvt(eeW[2], 128, 128, 128, 128)};
  const _Float16 *ebWt[4][3], *nbWt[4][3];
  for (int i = 0; i < 4; ++i) {
    ebWt[i][0] = cvt(ebW[i][0], 384, 128, 384, 128);
    ebWt[i][1] = cvt(ebW[i][1], 128, 128, 128, 128);
    ebWt[i][2] = cvt(ebW[i][2], 128, 128, 128, 128);
    nbWt[i][0] = cvt(nbW[i][0], 256, 128, 256, 128);
    nbWt[i][1] = cvt(nbW[i][1], 128, 128, 128, 128);
    nbWt[i][2] = cvt(nbW[i][2], 128, 128, 128, 128);
  }
  const _Float16 *dWt[3] = {cvt(dW[0], 128, 128, 128, 128),
                            cvt(dW[1], 128, 128, 128, 128),
                            cvt(dW[2], 128, 3, 128, 16)};

  const int tb = NWAVES * 32;
  auto grid = [&](int rows) { return ((rows + 15) / 16 + NWAVES - 1) / NWAVES; };

  // encoders
  k_mlp<M_ENC><<<grid(n_nodes), tb, 0, stream>>>(
      nodes, nullptr, nullptr, nullptr, neWt[0], neWt[1], neWt[2],
      neB[0], neB[1], neB[2], neG, neBeta, hn, n_nodes);
  k_mlp<M_ENC><<<grid(n_edges), tb, 0, stream>>>(
      edges, nullptr, nullptr, nullptr, eeWt[0], eeWt[1], eeWt[2],
      eeB[0], eeB[1], eeB[2], eeG, eeBeta, he, n_edges);

  // message-passing layers
  for (int i = 0; i < 4; ++i) {
    k_mlp<M_EBLK><<<grid(n_edges), tb, 0, stream>>>(
        hn, he, senders, receivers, ebWt[i][0], ebWt[i][1], ebWt[i][2],
        ebB[i][0], ebB[i][1], ebB[i][2], ebG[i], ebBeta[i], he, n_edges);

    int zt = n_nodes * HID;
    k_zero<<<(zt + 255) / 256, 256, 0, stream>>>(agg, zt);
    int st = n_edges * HID;
    k_scatter<<<(st + 255) / 256, 256, 0, stream>>>(he, receivers, agg, n_edges);

    k_mlp<M_NBLK><<<grid(n_nodes), tb, 0, stream>>>(
        hn, agg, nullptr, nullptr, nbWt[i][0], nbWt[i][1], nbWt[i][2],
        nbB[i][0], nbB[i][1], nbB[i][2], nbG[i], nbBeta[i], hn, n_nodes);
  }

  // decoder
  k_mlp<M_DEC><<<grid(n_nodes), tb, 0, stream>>>(
      hn, nullptr, nullptr, nullptr, dWt[0], dWt[1], dWt[2],
      dB[0], dB[1], dB[2], nullptr, nullptr, (float*)d_out, n_nodes);
}
